// STFT_7782480740689
// MI455X (gfx1250) — matrix operand order repaired and checked by reading, not compiled, not run on hardware
//
#include <hip/hip_runtime.h>
#include <cstdint>
#include <cstddef>

typedef __attribute__((ext_vector_type(16))) _Float16 v16h;
typedef __attribute__((ext_vector_type(8)))  _Float16 v8h;
typedef __attribute__((ext_vector_type(4)))  _Float16 v4h;
typedef __attribute__((ext_vector_type(8)))  float    v8f;
typedef __attribute__((ext_vector_type(4)))  float    v4f;

namespace {
constexpr int SIG_LEN  = 524288;   // samples per batch row
constexpr int FILT_LEN = 1024;
constexpr int HOP      = 512;
constexpr int CUTOFF   = 513;      // real rows; imag rows start at 513
constexpr int NFRAMES  = 1027;
constexpr int NBATCH   = 64;

constexpr int TILE_M = 128;        // 64 real + 64 imag rows per block
constexpr int TILE_N = 64;         // frames per block (4 WMMA N-subtiles/wave)
constexpr int TILE_K = 32;         // K per WMMA step (f16 K=32)
constexpr int KSTEPS = FILT_LEN / TILE_K;

constexpr int A_STR = 48;          // f16 elems per LDS row (96B: 16B-aligned, conflict-skewed)
constexpr int B_STR = 48;
constexpr int C_STR = 65;          // f32 elems per LDS row for epilogue tile
}

union alignas(16) SMem {
    struct {
        _Float16 A[TILE_M * A_STR];   // basis tile, row-major [m][k]
        _Float16 B[TILE_N * B_STR];   // frame tile, TRANSPOSED [n][k]
    } ab;
    float C[TILE_M * C_STR];          // fp32 result tile for mag/phase fusion
};

struct TrueT  { static constexpr bool value = true;  };
struct FalseT { static constexpr bool value = false; };

__global__ __launch_bounds__(256)
void stft_wmma(const float* __restrict__ x,
               const float* __restrict__ w,
               float* __restrict__ out)
{
    __shared__ SMem sm;

    const int tid  = threadIdx.x;
    const int wave = tid >> 5;         // 8 waves of 32 (wave32)
    const int lane = tid & 31;

    const int t0 = blockIdx.x * TILE_N;   // frame tile base
    const int f0 = blockIdx.y * 64;       // frequency-pair tile base
    const int b  = blockIdx.z;

    const float* xb = x + (size_t)b * SIG_LEN;

    // ---- per-thread staging assignments (fixed across K loop) ----
    // A: 128 rows x 32 f32 = 4096 elems / 256 thr = 16 each (one half-row)
    const int  ar   = tid >> 1;                 // local A row 0..127
    const int  ak0  = (tid & 1) * 16;           // k sub-base 0 or 16
    const int  fidx = (ar < 64) ? (f0 + ar) : (f0 + ar - 64);
    const bool aValid = fidx < CUTOFF;
    const int  arow  = (ar < 64) ? fidx : (CUTOFF + fidx);  // real | imag basis row

    // B: 64 frames x 32 k = 2048 elems / 256 thr = 8 each (contiguous k run)
    const int  bt    = tid >> 2;                // local frame 0..63
    const int  bk0   = (tid & 3) * 8;
    const int  tg    = t0 + bt;                 // global frame
    const bool tValid = tg < NFRAMES;
    const int  bOff  = tg * HOP - FILT_LEN + bk0;   // + k0 each step; may be negative

    // ---- accumulators: wave owns M rows [wave*16,+16), N = 0..63 ----
    v8f acc[4] = {};

    // fragment gather offsets (ISA 7.12.2 layouts)
    const int m16 = lane & 15;
    const int kA  = (lane >> 4) * 8;    // A: K half-select per lane half
    const int kB  = (lane >> 4) * 8;    // B: K base per lane half

    const _Float16* Abase = &sm.ab.A[(wave * 16 + m16) * A_STR];

    auto kloop = [&](auto fast_tag) {
        constexpr bool FAST = decltype(fast_tag)::value;
        for (int kb = 0; kb < KSTEPS; ++kb) {
            const int k0 = kb * TILE_K;

            // ---------- stage A: basis f32 -> f16 ----------
            {
                union { _Float16 h[16]; v8h v[2]; } tmp;
                const float* src = w + arow * FILT_LEN + k0 + ak0;
                if (FAST || aValid) {
                    #pragma unroll
                    for (int j = 0; j < 16; j += 4) {
                        v4f v = *(const v4f*)(src + j);
                        tmp.h[j+0] = (_Float16)v[0];
                        tmp.h[j+1] = (_Float16)v[1];
                        tmp.h[j+2] = (_Float16)v[2];
                        tmp.h[j+3] = (_Float16)v[3];
                    }
                    __builtin_prefetch(src + TILE_K, 0, 0);  // next K-slab (global_prefetch_b8)
                } else {
                    #pragma unroll
                    for (int j = 0; j < 16; ++j) tmp.h[j] = (_Float16)0.0f;
                }
                v8h* dst = (v8h*)&sm.ab.A[ar * A_STR + ak0];
                dst[0] = tmp.v[0];
                dst[1] = tmp.v[1];
            }

            // ---------- stage B (transposed): frames f32 -> f16 ----------
            {
                union { _Float16 h[8]; v8h v; } tmp;
                const int p0 = bOff + k0;
                if (FAST) {
                    v4f v0 = *(const v4f*)(xb + p0);
                    v4f v1 = *(const v4f*)(xb + p0 + 4);
                    #pragma unroll
                    for (int j = 0; j < 4; ++j) {
                        tmp.h[j]     = (_Float16)v0[j];
                        tmp.h[4 + j] = (_Float16)v1[j];
                    }
                } else {
                    #pragma unroll
                    for (int j = 0; j < 8; ++j) {
                        const int p = p0 + j;
                        tmp.h[j] = (tValid && p >= 0 && p < SIG_LEN)
                                     ? (_Float16)xb[p] : (_Float16)0.0f;
                    }
                }
                *(v8h*)&sm.ab.B[bt * B_STR + bk0] = tmp.v;
            }

            __syncthreads();

            // ---------- WMMA fragments from LDS ----------
            union { v16h v; v8h h[2]; } afrag;
            afrag.h[0] = *(const v8h*)(Abase + kA);        // K = kA..kA+7
            afrag.h[1] = *(const v8h*)(Abase + 16 + kA);   // K = 16+kA..+7

            #pragma unroll
            for (int s = 0; s < 4; ++s) {
                union { v16h v; v8h h[2]; } bfrag;
                const _Float16* Bp = &sm.ab.B[(s * 16 + m16) * B_STR + kB];
                bfrag.h[0] = *(const v8h*)(Bp);
                bfrag.h[1] = *(const v8h*)(Bp + 16);
                acc[s] = __builtin_amdgcn_wmma_f32_16x16x32_f16(
                             false, afrag.v, false, bfrag.v, (short)0, acc[s],
                             false, false);
            }

            __syncthreads();   // protect LDS tiles before restage
        }
    };

    // Block-uniform specialization: interior frame tile (no padding touched)
    // and all basis rows valid -> straight-line staging, no exec-mask churn.
    const bool fast = (t0 >= 2) && (t0 + TILE_N - 1 <= 1024) && (f0 + 63 < CUTOFF);
    if (fast) kloop(TrueT{});
    else      kloop(FalseT{});

    // ================= epilogue: fuse magnitude + phase =================
    // C/D layout: VGPR v -> M = v + 8*(lane>=16), N = lane%16
    {
        const int mo = (lane >> 4) * 8;
        #pragma unroll
        for (int s = 0; s < 4; ++s)
            #pragma unroll
            for (int v = 0; v < 8; ++v)
                sm.C[(wave * 16 + mo + v) * C_STR + s * 16 + m16] = acc[s][v];
    }
    __syncthreads();

    // 64 freq rows x 64 frames; 16 contiguous frames per thread
    const int fl = tid >> 2;             // 0..63
    const int tq = (tid & 3) * 16;       // 0,16,32,48
    const int f  = f0 + fl;
    if (f < CUTOFF) {
        const size_t magBase  = ((size_t)b * CUTOFF + f) * NFRAMES;
        const size_t phaseOfs = (size_t)NBATCH * CUTOFF * NFRAMES;
        #pragma unroll
        for (int j = 0; j < 16; ++j) {
            const int t = t0 + tq + j;
            if (t < NFRAMES) {
                const float re = sm.C[fl * C_STR + tq + j];
                const float im = sm.C[(64 + fl) * C_STR + tq + j];
                out[magBase + t]            = sqrtf(re * re + im * im + 1e-10f);
                out[phaseOfs + magBase + t] = atan2f(im, re);
            }
        }
    }
}

extern "C" void kernel_launch(void* const* d_in, const int* in_sizes, int n_in,
                              void* d_out, int out_size, void* d_ws, size_t ws_size,
                              hipStream_t stream)
{
    (void)in_sizes; (void)n_in; (void)d_ws; (void)ws_size; (void)out_size;
    const float* x   = (const float*)d_in[0];   // (64, 524288) f32
    const float* w   = (const float*)d_in[1];   // (1026, 1024) f32 basis
    float*       out = (float*)d_out;           // mag (64,513,1027) ++ phase (64,513,1027)

    dim3 grid((NFRAMES + TILE_N - 1) / TILE_N,   // 17 frame tiles
              (CUTOFF + 63) / 64,                // 9 frequency-pair tiles
              NBATCH);                           // 64 batches
    stft_wmma<<<grid, 256, 0, stream>>>(x, w, out);
}
